// GPT2_29343216566700
// MI455X (gfx1250) — matrix operand order repaired and checked by reading, not compiled, not run on hardware
//
#include <hip/hip_runtime.h>
#include <hip/hip_bf16.h>
#include <math.h>

// ---------------------------------------------------------------------------
// GPT-2 small forward on MI455X (gfx1250, wave32, WMMA).
// GEMMs: v_wmma_f32_16x16x32_bf16, LDS-staged double-buffered K pipeline.
// Staging uses GLOBAL_LOAD_ASYNC_TO_LDS_B128 (ASYNCcnt) when the builtin is
// available, else global->VGPR->ds_store_b128.
// ---------------------------------------------------------------------------

typedef __attribute__((ext_vector_type(16))) __bf16 v16bf;
typedef __attribute__((ext_vector_type(8)))  __bf16 v8bf;
typedef __attribute__((ext_vector_type(8)))  float  v8f;
// payload type for the async-to-LDS b128 builtin (per compiler diagnostic:
// "__attribute__((__vector_size__(4 * sizeof(int)))) int *")
typedef int v4i_vs __attribute__((vector_size(4 * sizeof(int))));

#ifndef __has_builtin
#define __has_builtin(x) 0
#endif

#if __has_builtin(__builtin_amdgcn_global_load_async_to_lds_b128)
#define HAS_ASYNC_LDS 1
#else
#define HAS_ASYNC_LDS 0
#endif

#if __has_builtin(__builtin_amdgcn_s_wait_asynccnt)
#define WAIT_ASYNC(n) __builtin_amdgcn_s_wait_asynccnt(n)
#else
#define WAIT_ASYNC(n) asm volatile("s_wait_asynccnt %0" ::"i"(n) : "memory")
#endif

#define LNUM  12
#define HNUM  12
#define EDIM  768
#define VDIM  50257
#define TSEQ  1024
#define BNUM  2
#define DHEAD 64
#define MT    (BNUM * TSEQ)

// LDS tile: 128 rows x 32 k-halfs, padded to 40 halfs (80 B) per row so the
// 16-lane fragment reads step 20 banks/row -> conflict-free b128 accesses.
#define LROW 40

__device__ __forceinline__ float gelu_exact(float x) {
  return 0.5f * x * (1.0f + erff(x * 0.70710678118654752f));
}

#if HAS_ASYNC_LDS
// one 16-byte async copy global -> LDS (tracked by ASYNCcnt)
__device__ __forceinline__ void async_copy16(const __bf16* g, __bf16* l) {
  __builtin_amdgcn_global_load_async_to_lds_b128(
      (v4i_vs*)g, (v4i_vs*)l, 0, 0);
}
#endif

// ---------------------------------------------------------------------------
// fp32 -> bf16 elementwise conversion (weights, once per launch)
// ---------------------------------------------------------------------------
__global__ void f32_to_bf16_kernel(const float* __restrict__ src,
                                   __bf16* __restrict__ dst, size_t n) {
  size_t i = (size_t)blockIdx.x * blockDim.x + threadIdx.x;
  if (i < n) dst[i] = (__bf16)src[i];
}

// ---------------------------------------------------------------------------
// Token + position embedding: x[b,t,:] = wte[idx[b,t],:] + wpe[t,:]
// ---------------------------------------------------------------------------
__global__ void embed_kernel(const int* __restrict__ idx,
                             const float* __restrict__ wte,
                             const float* __restrict__ wpe,
                             float* __restrict__ x) {
  const size_t i = (size_t)blockIdx.x * blockDim.x + threadIdx.x;
  if (i >= (size_t)MT * EDIM) return;
  const int e  = (int)(i % EDIM);
  const int bt = (int)(i / EDIM);
  const int t  = bt & (TSEQ - 1);
  x[i] = wte[(size_t)idx[bt] * EDIM + e] + wpe[(size_t)t * EDIM + e];
}

// ---------------------------------------------------------------------------
// LayerNorm over E=768, writes bf16 (GEMM A operand). One block per row.
// ---------------------------------------------------------------------------
__global__ __launch_bounds__(256)
void layernorm_bf16_kernel(const float* __restrict__ x,
                           const float* __restrict__ g,
                           const float* __restrict__ b,
                           __bf16* __restrict__ out) {
  const int row = blockIdx.x;
  const float* xr = x + (size_t)row * EDIM;
  const int tid = threadIdx.x;
  __shared__ float s1[256], s2[256];
  float v[3];
  float sum = 0.f, sq = 0.f;
#pragma unroll
  for (int j = 0; j < 3; ++j) {
    float t = xr[tid + j * 256];
    v[j] = t; sum += t; sq += t * t;
  }
  s1[tid] = sum; s2[tid] = sq;
  __syncthreads();
  for (int s = 128; s > 0; s >>= 1) {
    if (tid < s) { s1[tid] += s1[tid + s]; s2[tid] += s2[tid + s]; }
    __syncthreads();
  }
  const float mean = s1[0] * (1.0f / EDIM);
  const float var  = s2[0] * (1.0f / EDIM) - mean * mean;
  const float rs = rsqrtf(var + 1e-5f);
  __bf16* orow = out + (size_t)row * EDIM;
#pragma unroll
  for (int j = 0; j < 3; ++j) {
    const int e = tid + j * 256;
    orow[e] = (__bf16)((v[j] - mean) * rs * g[e] + b[e]);
  }
}

// ---------------------------------------------------------------------------
// Gather V into [B,H,D,T] bf16 so P@V sees a K-contiguous B operand.
// ---------------------------------------------------------------------------
__global__ void vtrans_kernel(const __bf16* __restrict__ qkv,
                              __bf16* __restrict__ vT) {
  const size_t i = (size_t)blockIdx.x * blockDim.x + threadIdx.x;
  if (i >= (size_t)BNUM * HNUM * DHEAD * TSEQ) return;
  const int t = (int)(i & (TSEQ - 1));
  const size_t hd = i >> 10;
  const int d  = (int)(hd & (DHEAD - 1));
  const int bh = (int)(hd >> 6);
  const int b = bh / HNUM, h = bh % HNUM;
  vT[i] = qkv[((size_t)(b * TSEQ + t)) * (3 * EDIM) + 2 * EDIM + h * DHEAD + d];
}

// ---------------------------------------------------------------------------
// Causal softmax in-place on bf16 scores [BH, T, T]. One block per (bh,q).
// ---------------------------------------------------------------------------
__global__ __launch_bounds__(256)
void softmax_causal_kernel(__bf16* __restrict__ S) {
  const int r = blockIdx.x;            // r = bh*T + q
  const int q = r & (TSEQ - 1);
  __bf16* row = S + (size_t)r * TSEQ;
  const int tid = threadIdx.x;
  __shared__ float red[256];
  float vals[4];
  float mx = -3.0e38f;
#pragma unroll
  for (int j = 0; j < 4; ++j) {
    const int k = tid + j * 256;
    float v = (k <= q) ? (float)row[k] : -3.0e38f;
    vals[j] = v;
    mx = fmaxf(mx, v);
  }
  red[tid] = mx; __syncthreads();
  for (int s = 128; s > 0; s >>= 1) {
    if (tid < s) red[tid] = fmaxf(red[tid], red[tid + s]);
    __syncthreads();
  }
  mx = red[0];
  __syncthreads();
  float sum = 0.f;
#pragma unroll
  for (int j = 0; j < 4; ++j) {
    const int k = tid + j * 256;
    float e = (k <= q) ? __expf(vals[j] - mx) : 0.f;
    vals[j] = e;
    sum += e;
  }
  red[tid] = sum; __syncthreads();
  for (int s = 128; s > 0; s >>= 1) {
    if (tid < s) red[tid] += red[tid + s];
    __syncthreads();
  }
  const float inv = 1.0f / red[0];
#pragma unroll
  for (int j = 0; j < 4; ++j) {
    const int k = tid + j * 256;
    row[k] = (__bf16)(vals[j] * inv);
  }
}

// ---------------------------------------------------------------------------
// bf16 WMMA GEMM:  C[m,n] = scale * sum_k A[m,k] * Bw[n,k]  (+bias)(+resid)
// A row-major [M,K] (lda), Bw row-major [N,K] (ldb) == B^T, K-contiguous.
// 128x128 block tile, 256 threads = 8 wave32 waves (4x2 grid), each wave
// 32x64 = 2x4 WMMA 16x16 tiles. K stepped by 32 through a double-buffered
// LDS stage (async-to-LDS when available). Batched over blockIdx.z.
// causalSkip : drop fully-masked upper-triangular blocks (Q.K^T)
// causalKLim : limit K loop to blockM+128 (P.V: probs beyond q are zero)
// ---------------------------------------------------------------------------
template <int OUT_BF16, int DO_GELU>
__global__ __launch_bounds__(256)
void gemm_wmma_bf16(const __bf16* __restrict__ A, int lda,
                    long long strideAb, long long strideAh,
                    const __bf16* __restrict__ Bw, int ldb,
                    long long strideBb, long long strideBh,
                    void* __restrict__ Cout, int ldc,
                    long long strideCb, long long strideCh,
                    const float* __restrict__ bias,
                    const float* __restrict__ resid,
                    int M, int N, int K, float scale,
                    int hdiv, int causalSkip, int causalKLim) {
  const int blockM = blockIdx.y * 128;
  const int blockN = blockIdx.x * 128;
  if (causalSkip && blockN > blockM) return;  // whole block above diagonal

  const int z  = blockIdx.z;
  const int zb = z / hdiv, zh = z % hdiv;
  A  += (size_t)zb * strideAb + (size_t)zh * strideAh;
  Bw += (size_t)zb * strideBb + (size_t)zh * strideBh;
  const size_t coutOff = (size_t)zb * strideCb + (size_t)zh * strideCh;

  __shared__ __bf16 smA[2][128 * LROW];
  __shared__ __bf16 smB[2][128 * LROW];

  const int tid  = threadIdx.x;
  const int wave = tid >> 5, lane = tid & 31;
  const int wm = wave >> 1, wn = wave & 1;   // 4x2 wave grid
  const int m16 = lane & 15, sel = lane >> 4;

  // ---- cooperative staging assignment: thread -> one 32B (16-half) chunk
  const int srow  = tid >> 1;   // 0..127 : A row / B col within the tile
  const int shalf = tid & 1;    // which 16-half chunk of the 32-half row
  int bcol = blockN + srow;
  if (bcol >= N) bcol = N - 1;  // ragged N (LM head): duplicate, store guarded
  const __bf16* gA = A  + (size_t)(blockM + srow) * lda + shalf * 16;
  const __bf16* gB = Bw + (size_t)bcol * ldb + shalf * 16;
  const int sOff = srow * LROW + shalf * 16;

  int Kend = K;
  if (causalKLim) { int kl = blockM + 128; Kend = kl < K ? kl : K; }
  const int nk = Kend >> 5;

  // issue one K-step tile (A+B) into LDS buffer `buf`
  auto stage = [&](int k0, int buf) {
    __builtin_prefetch(gB + k0 + 64, 0, 0);  // speculative, next-next tile
#if HAS_ASYNC_LDS
    async_copy16(gA + k0,     &smA[buf][sOff]);
    async_copy16(gA + k0 + 8, &smA[buf][sOff + 8]);
    async_copy16(gB + k0,     &smB[buf][sOff]);
    async_copy16(gB + k0 + 8, &smB[buf][sOff + 8]);
#else
    v8bf a0 = *(const v8bf*)(gA + k0);
    v8bf a1 = *(const v8bf*)(gA + k0 + 8);
    v8bf b0 = *(const v8bf*)(gB + k0);
    v8bf b1 = *(const v8bf*)(gB + k0 + 8);
    *(v8bf*)&smA[buf][sOff]     = a0;
    *(v8bf*)&smA[buf][sOff + 8] = a1;
    *(v8bf*)&smB[buf][sOff]     = b0;
    *(v8bf*)&smB[buf][sOff + 8] = b1;
#endif
  };

  v8f acc[2][4];
#pragma unroll
  for (int mi = 0; mi < 2; ++mi)
#pragma unroll
    for (int ni = 0; ni < 4; ++ni)
#pragma unroll
      for (int r = 0; r < 8; ++r) acc[mi][ni][r] = 0.f;

  stage(0, 0);
  int buf = 0;
  for (int kt = 0; kt < nk; ++kt) {
    const bool more = (kt + 1 < nk);
    if (more) stage((kt + 1) << 5, buf ^ 1);   // overlap with compute
#if HAS_ASYNC_LDS
    if (more) WAIT_ASYNC(4);   // tile kt's 4 per-wave loads done (in-order)
    else      WAIT_ASYNC(0);
#endif
    __syncthreads();           // tile kt visible to all waves

    v16bf afrag[2], bfrag[4];
    // A fragment 16x32: lane<16 K=0..7(V0-3),16..23(V4-7); lane>=16 +8
#pragma unroll
    for (int mi = 0; mi < 2; ++mi) {
      const __bf16* ap = &smA[buf][(wm * 32 + mi * 16 + m16) * LROW + sel * 8];
      v8bf lo = *(const v8bf*)ap;
      v8bf hi = *(const v8bf*)(ap + 16);
#pragma unroll
      for (int i = 0; i < 8; ++i) { afrag[mi][i] = lo[i]; afrag[mi][i + 8] = hi[i]; }
    }
    // B fragment 32x16: lane<16 K=0..7,16..23 of col n=lane%16; lane>=16 +8
#pragma unroll
    for (int ni = 0; ni < 4; ++ni) {
      const __bf16* bp = &smB[buf][(wn * 64 + ni * 16 + m16) * LROW + sel * 8];
      v8bf lo = *(const v8bf*)bp;
      v8bf hi = *(const v8bf*)(bp + 16);
#pragma unroll
      for (int i = 0; i < 8; ++i) { bfrag[ni][i] = lo[i]; bfrag[ni][i + 8] = hi[i]; }
    }
#pragma unroll
    for (int mi = 0; mi < 2; ++mi)
#pragma unroll
      for (int ni = 0; ni < 4; ++ni)
        acc[mi][ni] = __builtin_amdgcn_wmma_f32_16x16x32_bf16(
            false, afrag[mi], false, bfrag[ni], (short)0, acc[mi][ni],
            false, false);

    __syncthreads();           // all waves done reading buf before overwrite
    buf ^= 1;
  }

  // Epilogue. C/D layout: VGPR r -> row = sel*8 + r, col = lane%16.
#pragma unroll
  for (int mi = 0; mi < 2; ++mi) {
#pragma unroll
    for (int ni = 0; ni < 4; ++ni) {
      const int col = blockN + wn * 64 + ni * 16 + m16;
      if (col >= N) continue;
      const float bv = bias ? bias[col] : 0.f;
#pragma unroll
      for (int r = 0; r < 8; ++r) {
        const int row = blockM + wm * 32 + mi * 16 + sel * 8 + r;
        float v = acc[mi][ni][r] * scale + bv;
        if (DO_GELU) v = gelu_exact(v);
        const size_t co = coutOff + (size_t)row * ldc + col;
        if (OUT_BF16) {
          ((__bf16*)Cout)[co] = (__bf16)v;
        } else {
          if (resid) v += resid[co];
          ((float*)Cout)[co] = v;
        }
      }
    }
  }
}

// ---------------------------------------------------------------------------
// Host orchestration
// ---------------------------------------------------------------------------
extern "C" void kernel_launch(void* const* d_in, const int* in_sizes, int n_in,
                              void* d_out, int out_size, void* d_ws,
                              size_t ws_size, hipStream_t stream) {
  (void)in_sizes; (void)n_in; (void)out_size; (void)ws_size;

  const int*   idx    = (const int*)  d_in[0];
  const float* wte    = (const float*)d_in[1];
  const float* wpe    = (const float*)d_in[2];
  const float* head_w = (const float*)d_in[3];
  const float* lnf_g  = (const float*)d_in[4];
  const float* lnf_b  = (const float*)d_in[5];
  const float* ln1_g  = (const float*)d_in[6];
  const float* ln1_b  = (const float*)d_in[7];
  const float* ln2_g  = (const float*)d_in[8];
  const float* ln2_b  = (const float*)d_in[9];
  const float* attn_w = (const float*)d_in[10];
  const float* attn_b = (const float*)d_in[11];
  const float* proj_w = (const float*)d_in[12];
  const float* proj_b = (const float*)d_in[13];
  const float* fc_w   = (const float*)d_in[14];
  const float* fc_b   = (const float*)d_in[15];
  const float* fc2_w  = (const float*)d_in[16];
  const float* fc2_b  = (const float*)d_in[17];

  char* ws = (char*)d_ws;
  size_t off = 0;
  auto give = [&](size_t bytes) -> char* {
    char* p = ws + off;
    off = (off + bytes + 255) & ~(size_t)255;
    return p;
  };

  const size_t n_attn = (size_t)LNUM * 3 * EDIM * EDIM;
  const size_t n_proj = (size_t)LNUM * EDIM * EDIM;
  const size_t n_fc   = (size_t)LNUM * 4 * EDIM * EDIM;
  const size_t n_fc2  = (size_t)LNUM * EDIM * 4 * EDIM;
  const size_t n_head = (size_t)VDIM * EDIM;

  __bf16* attn_w_bf = (__bf16*)give(n_attn * 2);
  __bf16* proj_w_bf = (__bf16*)give(n_proj * 2);
  __bf16* fc_w_bf   = (__bf16*)give(n_fc * 2);
  __bf16* fc2_w_bf  = (__bf16*)give(n_fc2 * 2);
  __bf16* head_w_bf = (__bf16*)give(n_head * 2);
  float*  x     = (float*) give((size_t)MT * EDIM * 4);
  __bf16* hbf   = (__bf16*)give((size_t)MT * EDIM * 2);
  __bf16* qkvbf = (__bf16*)give((size_t)MT * 3 * EDIM * 2);
  __bf16* vT    = (__bf16*)give((size_t)BNUM * HNUM * DHEAD * TSEQ * 2);
  __bf16* S     = (__bf16*)give((size_t)BNUM * HNUM * TSEQ * TSEQ * 2);
  __bf16* ybf   = (__bf16*)give((size_t)MT * EDIM * 2);
  __bf16* mbf   = (__bf16*)give((size_t)MT * 4 * EDIM * 2);

  auto conv = [&](const float* s, __bf16* d, size_t n) {
    int blocks = (int)((n + 255) / 256);
    f32_to_bf16_kernel<<<blocks, 256, 0, stream>>>(s, d, n);
  };
  conv(attn_w, attn_w_bf, n_attn);
  conv(proj_w, proj_w_bf, n_proj);
  conv(fc_w,   fc_w_bf,   n_fc);
  conv(fc2_w,  fc2_w_bf,  n_fc2);
  conv(head_w, head_w_bf, n_head);

  // Embedding
  {
    size_t n = (size_t)MT * EDIM;
    embed_kernel<<<(int)((n + 255) / 256), 256, 0, stream>>>(idx, wte, wpe, x);
  }

  const long long TT  = (long long)TSEQ * TSEQ;
  const long long QKS = (long long)TSEQ * 3 * EDIM;
  const float attn_scale = 0.125f;  // 1/sqrt(64)

  for (int l = 0; l < LNUM; ++l) {
    // LN1 -> hbf
    layernorm_bf16_kernel<<<MT, 256, 0, stream>>>(
        x, ln1_g + (size_t)l * EDIM, ln1_b + (size_t)l * EDIM, hbf);

    // qkv = hbf @ attn_w[l]^T + attn_b[l]   -> bf16 [MT, 3E]
    gemm_wmma_bf16<1, 0><<<dim3(3 * EDIM / 128, MT / 128, 1), 256, 0, stream>>>(
        hbf, EDIM, 0, 0,
        attn_w_bf + (size_t)l * 3 * EDIM * EDIM, EDIM, 0, 0,
        qkvbf, 3 * EDIM, 0, 0,
        attn_b + (size_t)l * 3 * EDIM, nullptr,
        MT, 3 * EDIM, EDIM, 1.0f, 1, 0, 0);

    // V -> [B,H,D,T]
    {
      size_t n = (size_t)BNUM * HNUM * DHEAD * TSEQ;
      vtrans_kernel<<<(int)((n + 255) / 256), 256, 0, stream>>>(qkvbf, vT);
    }

    // S = scale * Q @ K^T  per (b,h), causal block-skip  -> bf16 [BH,T,T]
    gemm_wmma_bf16<1, 0><<<dim3(TSEQ / 128, TSEQ / 128, BNUM * HNUM), 256, 0, stream>>>(
        qkvbf, 3 * EDIM, QKS, DHEAD,
        qkvbf + EDIM, 3 * EDIM, QKS, DHEAD,
        S, TSEQ, (long long)HNUM * TT, TT,
        nullptr, nullptr,
        TSEQ, TSEQ, DHEAD, attn_scale, HNUM, /*causalSkip=*/1, 0);

    // softmax (causal, in-place)
    softmax_causal_kernel<<<BNUM * HNUM * TSEQ, 256, 0, stream>>>(S);

    // Y = P @ V  per (b,h)  -> bf16 [B,T,E] (heads interleaved)
    gemm_wmma_bf16<1, 0><<<dim3(1, TSEQ / 128, BNUM * HNUM), 256, 0, stream>>>(
        S, TSEQ, (long long)HNUM * TT, TT,
        vT, TSEQ, (long long)HNUM * DHEAD * TSEQ, (long long)DHEAD * TSEQ,
        ybf, EDIM, (long long)TSEQ * EDIM, DHEAD,
        nullptr, nullptr,
        TSEQ, DHEAD, TSEQ, 1.0f, HNUM, 0, /*causalKLim=*/1);

    // x += Y @ proj_w[l]^T + proj_b[l]
    gemm_wmma_bf16<0, 0><<<dim3(EDIM / 128, MT / 128, 1), 256, 0, stream>>>(
        ybf, EDIM, 0, 0,
        proj_w_bf + (size_t)l * EDIM * EDIM, EDIM, 0, 0,
        x, EDIM, 0, 0,
        proj_b + (size_t)l * EDIM, x,
        MT, EDIM, EDIM, 1.0f, 1, 0, 0);

    // LN2 -> hbf
    layernorm_bf16_kernel<<<MT, 256, 0, stream>>>(
        x, ln2_g + (size_t)l * EDIM, ln2_b + (size_t)l * EDIM, hbf);

    // m = gelu(hbf @ fc_w[l]^T + fc_b[l]) -> bf16 [MT, 4E]
    gemm_wmma_bf16<1, 1><<<dim3(4 * EDIM / 128, MT / 128, 1), 256, 0, stream>>>(
        hbf, EDIM, 0, 0,
        fc_w_bf + (size_t)l * 4 * EDIM * EDIM, EDIM, 0, 0,
        mbf, 4 * EDIM, 0, 0,
        fc_b + (size_t)l * 4 * EDIM, nullptr,
        MT, 4 * EDIM, EDIM, 1.0f, 1, 0, 0);

    // x += m @ fc2_w[l]^T + fc2_b[l]
    gemm_wmma_bf16<0, 0><<<dim3(EDIM / 128, MT / 128, 1), 256, 0, stream>>>(
        mbf, 4 * EDIM, 0, 0,
        fc2_w_bf + (size_t)l * EDIM * 4 * EDIM, 4 * EDIM, 0, 0,
        x, EDIM, 0, 0,
        fc2_b + (size_t)l * EDIM, x,
        MT, EDIM, 4 * EDIM, 1.0f, 1, 0, 0);
  }

  // Final LN -> hbf
  layernorm_bf16_kernel<<<MT, 256, 0, stream>>>(x, lnf_g, lnf_b, hbf);

  // logits = hbf @ head_w^T  -> fp32 [MT, V]  (ragged N handled by clamp/guard)
  gemm_wmma_bf16<0, 0><<<dim3((VDIM + 127) / 128, MT / 128, 1), 256, 0, stream>>>(
      hbf, EDIM, 0, 0,
      head_w_bf, EDIM, 0, 0,
      (float*)d_out, VDIM, 0, 0,
      nullptr, nullptr,
      MT, VDIM, EDIM, 1.0f, 1, 0, 0);
}
